// SelfAttention_28381143892253
// MI455X (gfx1250) — matrix operand order repaired and checked by reading, not compiled, not run on hardware
//
#include <hip/hip_runtime.h>
#include <hip/hip_bf16.h>

typedef __attribute__((ext_vector_type(16))) _Float16 v16h;
typedef __attribute__((ext_vector_type(8)))  _Float16 v8h;
typedef __attribute__((ext_vector_type(8)))  float    v8f;
typedef __attribute__((ext_vector_type(4)))  int      v4i;

#define HEADS    16
#define HDIM     64
#define EMBED    1024

// ---- CDNA5 async global->LDS path (guarded; falls back to sync copy) ------
#if defined(__has_builtin)
#if __has_builtin(__builtin_amdgcn_global_load_async_to_lds_b128)
#define ASYNC_LDS 1
#endif
#endif
#ifndef ASYNC_LDS
#define ASYNC_LDS 0
#endif

__device__ __forceinline__ void async_copy16(const _Float16* gsrc, _Float16* ldst) {
#if ASYNC_LDS
    __builtin_amdgcn_global_load_async_to_lds_b128(
        (v4i*)gsrc, (v4i*)ldst, /*offset=*/0, /*cpol=*/0);
#else
    *(v8h*)ldst = *(const v8h*)gsrc;
#endif
}

__device__ __forceinline__ void async_wait0() {
#if ASYNC_LDS
#if __has_builtin(__builtin_amdgcn_s_wait_asynccnt)
    __builtin_amdgcn_s_wait_asynccnt(0);
#else
    asm volatile("s_wait_asynccnt 0" ::: "memory");
#endif
#endif
}

// ---------------------------------------------------------------------------
// Fragment index helpers (CDNA5 ISA 7.12.2 VGPR layouts, wave32)
// ---------------------------------------------------------------------------
__device__ __forceinline__ int a_kidx(int v, int e, int hi) {
    return ((v >> 2) << 4) + (hi << 3) + ((v & 3) << 1) + e;
}
__device__ __forceinline__ int b_kidx(int v, int e, int hi) {
    return ((v >> 2) << 4) + (hi << 3) + ((v & 3) << 1) + e;
}

__device__ __forceinline__ v8f wmma_f16(v16h a, v16h b, v8f c) {
    return __builtin_amdgcn_wmma_f32_16x16x32_f16(
        false, a, false, b, (short)0, c, false, false);
}

// ---- DPP16 row reductions: pure VALU, no LDS round trips -----------------
// quad_perm(0xB1)=swap pairs, quad_perm(0x4E)=swap quads-in-4,
// 0x141=row_half_mirror (combines quads), 0x140=row_mirror (combines octets)
template <int CTRL>
__device__ __forceinline__ float dpp_bcast(float x) {
    int r = __builtin_amdgcn_update_dpp(0, __builtin_bit_cast(int, x),
                                        CTRL, 0xf, 0xf, true);
    return __builtin_bit_cast(float, r);
}
__device__ __forceinline__ float rowmax16(float v) {
    v = fmaxf(v, dpp_bcast<0x0B1>(v));
    v = fmaxf(v, dpp_bcast<0x04E>(v));
    v = fmaxf(v, dpp_bcast<0x141>(v));
    v = fmaxf(v, dpp_bcast<0x140>(v));
    return v;
}
__device__ __forceinline__ float rowsum16(float v) {
    v += dpp_bcast<0x0B1>(v);
    v += dpp_bcast<0x04E>(v);
    v += dpp_bcast<0x141>(v);
    v += dpp_bcast<0x140>(v);
    return v;
}

// ---------------------------------------------------------------------------
// Kernel 1: per-head linear projection   X_head @ W^T + b
// ---------------------------------------------------------------------------
template <bool VT>
__global__ __launch_bounds__(256, 1)
void qkv_proj_kernel(const float* __restrict__ X,
                     const float* __restrict__ W,
                     const float* __restrict__ bias,
                     _Float16* __restrict__ dst,
                     int N, int L) {
    const int lane = threadIdx.x & 31;
    const int wave = threadIdx.x >> 5;
    const int tile = blockIdx.x * 8 + wave;
    const int row0 = tile << 4;                 // row among N*L*H
    const int m  = lane & 15;
    const int hi = lane >> 4;

    int r   = row0 + m;
    int n   = r / (L * HEADS);
    int rem = r - n * (L * HEADS);
    int l   = rem / HEADS;
    int h   = rem - l * HEADS;
    const float* src = X + (size_t)(n * L + l) * EMBED + h * HDIM;

    v16h a0, a1;
#pragma unroll
    for (int v = 0; v < 8; ++v)
#pragma unroll
        for (int e = 0; e < 2; ++e) {
            int k = a_kidx(v, e, hi);
            a0[2 * v + e] = (_Float16)src[k];
            a1[2 * v + e] = (_Float16)src[32 + k];
        }

#pragma unroll
    for (int et = 0; et < 4; ++et) {
        const int ecol = et * 16 + m;
        v16h b0, b1;
#pragma unroll
        for (int v = 0; v < 8; ++v)
#pragma unroll
            for (int e = 0; e < 2; ++e) {
                int kk = b_kidx(v, e, hi);
                b0[2 * v + e] = (_Float16)W[ecol * HDIM + kk];   // B[d][e]=W[e][d]
                b1[2 * v + e] = (_Float16)W[ecol * HDIM + 32 + kk];
            }
        v8f acc = {};
        acc = wmma_f16(a0, b0, acc);
        acc = wmma_f16(a1, b1, acc);
        const float bb = bias[ecol];
#pragma unroll
        for (int v = 0; v < 8; ++v) {
            int rr   = row0 + v + hi * 8;
            int nn   = rr / (L * HEADS);
            int rem2 = rr - nn * (L * HEADS);
            int ll   = rem2 / HEADS;
            int hh   = rem2 - ll * HEADS;
            const _Float16 val = (_Float16)(acc[v] + bb);
            if (VT)
                dst[((size_t)((nn * HEADS + hh) * HDIM + ecol)) * L + ll] = val;
            else
                dst[((size_t)((nn * HEADS + hh) * L) + ll) * HDIM + ecol] = val;
        }
    }
}

// ---------------------------------------------------------------------------
// Kernel 2: flash attention.  Block = 8 waves = 8 consecutive 16-row query
// tiles of the SAME (n, h).  K tiles staged into LDS via async global->LDS
// DMA, double buffered.  V read from transposed layout (contiguous per lane).
// ---------------------------------------------------------------------------
#define KPAD 72   // 144B row stride: 16 rows hit 16 distinct 256B-period slots
#define PPAD 40   // 80B row stride for the P transpose buffer

__global__ __launch_bounds__(256, 1)
void attn_kernel(const _Float16* __restrict__ Qp,
                 const _Float16* __restrict__ Kp,
                 const _Float16* __restrict__ Vt,
                 const int* __restrict__ mask,
                 _Float16* __restrict__ attn_out,
                 int N, int L) {
    __shared__ _Float16 sK[2][32][KPAD];         // ping-pong K tiles (~9 KB)
    __shared__ _Float16 sP[8][16 * PPAD];        // per-wave P staging (10 KB)

    const int lane   = threadIdx.x & 31;
    const int wave   = threadIdx.x >> 5;
    const int gid    = blockIdx.x * 8 + wave;
    const int qtiles = L >> 4;
    const int qt = gid % qtiles;
    const int h  = (gid / qtiles) % HEADS;
    const int n  = gid / (qtiles * HEADS);
    const int q0 = qt << 4;
    const int m  = lane & 15;
    const int hi = lane >> 4;

    const size_t headBase = (size_t)(n * HEADS + h) * L * HDIM;

    // per-thread K-tile staging slot: 256 threads x 8 halves = 32x64 tile
    const int srow = threadIdx.x >> 3;           // 0..31
    const int scol = (threadIdx.x & 7) * 8;      // 0,8,...,56
    const _Float16* kstage = Kp + headBase + (size_t)srow * HDIM + scol;

    // Q fragments (persist across the whole key loop)
    const _Float16* qrow = Qp + headBase + (size_t)(q0 + m) * HDIM;
    v16h aq0, aq1;
#pragma unroll
    for (int v = 0; v < 8; ++v)
#pragma unroll
        for (int e = 0; e < 2; ++e) {
            int k = a_kidx(v, e, hi);
            aq0[2 * v + e] = qrow[k];
            aq1[2 * v + e] = qrow[32 + k];
        }

    float mcur[8], lsum[8];
    v8f o[4] = {};
#pragma unroll
    for (int v = 0; v < 8; ++v) { mcur[v] = -3.0e38f; lsum[v] = 0.0f; }

    const float scale = 0.03125f;      // 1/sqrt(EMBED) = 1/32
    const float NEGS  = -3.125e18f;    // (-1e20)/32 : mask applied BEFORE scale
    const int*  mbase = mask + (size_t)n * L * L;
    _Float16*   P     = sP[wave];

    const int ktiles = L >> 5;

    // prime the pipeline: stage tile 0 into buffer 0
    async_copy16(kstage, &sK[0][srow][scol]);

    int buf = 0;
    for (int kt = 0; kt < ktiles; ++kt) {
        const int kb = kt << 5;

        async_wait0();
        __syncthreads();                         // sK[buf] ready; prior reads done
        if (kt + 1 < ktiles)                     // overlap next DMA with compute
            async_copy16(kstage + (size_t)(kb + 32) * HDIM, &sK[buf ^ 1][srow][scol]);

        // prefetch next mask rows, near-cache temporal
        __builtin_prefetch(mbase + (size_t)(q0 + m) * L + kb + 32, 0, 3);

        // ---- S = Q . K^T for 32 keys (two 16-col C tiles) ----
        v8f S[2];
#pragma unroll
        for (int ct = 0; ct < 2; ++ct) {
            const _Float16* krow = &sK[buf][ct * 16 + m][0];
            v16h b0, b1;
#pragma unroll
            for (int v = 0; v < 8; ++v)
#pragma unroll
                for (int e = 0; e < 2; ++e) {
                    int kk = b_kidx(v, e, hi);
                    b0[2 * v + e] = krow[kk];          // B[d][key] = K[key][d]
                    b1[2 * v + e] = krow[32 + kk];
                }
            v8f s = {};
            s = wmma_f16(aq0, b0, s);
            s = wmma_f16(aq1, b1, s);
            S[ct] = s;
        }

        // ---- mask (before scale, like reference), scale, online softmax ----
        float corr[8];
#pragma unroll
        for (int v = 0; v < 8; ++v) {
            const int qr = q0 + v + hi * 8;
            const int mv0 = mbase[(size_t)qr * L + kb + m];
            const int mv1 = mbase[(size_t)qr * L + kb + 16 + m];
            float s0 = mv0 ? S[0][v] * scale : NEGS;
            float s1 = mv1 ? S[1][v] * scale : NEGS;

            float rm = rowmax16(fmaxf(s0, s1));
            float mn = fmaxf(mcur[v], rm);
            float c  = __expf(mcur[v] - mn);
            float p0 = __expf(s0 - mn);
            float p1 = __expf(s1 - mn);
            lsum[v]  = lsum[v] * c + rowsum16(p0 + p1);
            mcur[v]  = mn;
            corr[v]  = c;

            const int rr = v + hi * 8;                 // stage P (C layout)
            P[rr * PPAD + m]      = (_Float16)p0;
            P[rr * PPAD + 16 + m] = (_Float16)p1;
        }

        // rescale running O by the softmax correction
#pragma unroll
        for (int dt = 0; dt < 4; ++dt)
#pragma unroll
            for (int v = 0; v < 8; ++v) o[dt][v] *= corr[v];

        // DS ops are in-order per wave; fence the cross-lane LDS transpose
        asm volatile("s_wait_dscnt 0" ::: "memory");

        // reload P in A layout (16x32)
        v16h ap;
#pragma unroll
        for (int v = 0; v < 8; ++v)
#pragma unroll
            for (int e = 0; e < 2; ++e)
                ap[2 * v + e] = P[m * PPAD + a_kidx(v, e, hi)];

        // ---- O += P . V  (V transposed: per-lane contiguous columns) ----
#pragma unroll
        for (int dt = 0; dt < 4; ++dt) {
            const _Float16* vcol =
                Vt + ((size_t)((n * HEADS + h) * HDIM + dt * 16 + m)) * L + kb;
            v16h bv;
#pragma unroll
            for (int v = 0; v < 8; ++v)
#pragma unroll
                for (int e = 0; e < 2; ++e)
                    bv[2 * v + e] = vcol[b_kidx(v, e, hi)];   // V[kb+kk][dcol]
            o[dt] = wmma_f16(ap, bv, o[dt]);
        }

        buf ^= 1;
    }

    // ---- normalize and store [N][L][EMBED] as f16 for the output GEMM ----
#pragma unroll
    for (int v = 0; v < 8; ++v) {
        const float inv = 1.0f / lsum[v];
        const int   qr  = q0 + v + hi * 8;
#pragma unroll
        for (int dt = 0; dt < 4; ++dt)
            attn_out[(size_t)(n * L + qr) * EMBED + h * HDIM + dt * 16 + m] =
                (_Float16)(o[dt][v] * inv);
    }
}

// ---------------------------------------------------------------------------
// Kernel 3: out = A @ Wo^T + bo.   A: [R][1024] f16, Wo: [1024][1024] f32.
// ---------------------------------------------------------------------------
#define WPAD 40

__global__ __launch_bounds__(256, 1)
void out_proj_kernel(const _Float16* __restrict__ A,
                     const float* __restrict__ Wo,
                     const float* __restrict__ bo,
                     float* __restrict__ out) {
    __shared__ _Float16 sW[64][WPAD];            // [ecol][kk] (5 KB)

    const int lane = threadIdx.x & 31;
    const int wave = threadIdx.x >> 5;
    const int e0 = blockIdx.x * 64;
    const int r0 = blockIdx.y * 128 + wave * 16;
    const int m  = lane & 15;
    const int hi = lane >> 4;

    const _Float16* arow = A + (size_t)(r0 + m) * EMBED;

    v8f acc[4] = {};
    for (int kc = 0; kc < EMBED; kc += 32) {
        __syncthreads();
        {   // cooperative f32->f16 stage of Wo[e0..e0+63][kc..kc+31]
            const int t  = threadIdx.x;
            const int ec = t >> 2;               // 0..63
            const int k8 = (t & 3) * 8;          // 0,8,16,24
            const float* wr = Wo + (size_t)(e0 + ec) * EMBED + kc + k8;
#pragma unroll
            for (int j = 0; j < 8; ++j) sW[ec][k8 + j] = (_Float16)wr[j];
        }
        __syncthreads();

        v16h a;
#pragma unroll
        for (int v = 0; v < 8; ++v)
#pragma unroll
            for (int e = 0; e < 2; ++e)
                a[2 * v + e] = arow[kc + a_kidx(v, e, hi)];

#pragma unroll
        for (int et = 0; et < 4; ++et) {
            v16h b;
#pragma unroll
            for (int v = 0; v < 8; ++v)
#pragma unroll
                for (int e = 0; e < 2; ++e)
                    b[2 * v + e] = sW[et * 16 + m][b_kidx(v, e, hi)];
            acc[et] = wmma_f16(a, b, acc[et]);
        }
    }

#pragma unroll
    for (int et = 0; et < 4; ++et) {
        const int col = e0 + et * 16 + m;
        const float bb = bo[col];
#pragma unroll
        for (int v = 0; v < 8; ++v)
            out[(size_t)(r0 + v + hi * 8) * EMBED + col] = acc[et][v] + bb;
    }
}

// ---------------------------------------------------------------------------
extern "C" void kernel_launch(void* const* d_in, const int* in_sizes, int n_in,
                              void* d_out, int out_size, void* d_ws, size_t ws_size,
                              hipStream_t stream) {
    const float* query = (const float*)d_in[0];
    const float* key   = (const float*)d_in[1];
    const float* value = (const float*)d_in[2];
    const int*   mask  = (const int*)d_in[3];
    const float* Wq = (const float*)d_in[4];
    const float* bq = (const float*)d_in[5];
    const float* Wk = (const float*)d_in[6];
    const float* bk = (const float*)d_in[7];
    const float* Wv = (const float*)d_in[8];
    const float* bv = (const float*)d_in[9];
    const float* Wo = (const float*)d_in[10];
    const float* bo = (const float*)d_in[11];
    float* out = (float*)d_out;

    // derive N, L:  |query| = N*L*1024, |mask| = N*L*L
    const long long qs = in_sizes[0];
    const long long ms = in_sizes[3];
    const int L = (int)((1024LL * ms) / qs);
    const int N = (int)(qs / ((long long)L * EMBED));

    const size_t per = (size_t)N * L * EMBED;    // f16 elements per tensor
    _Float16* Qp = (_Float16*)d_ws;
    _Float16* Kp = Qp + per;
    _Float16* Vt = Kp + per;                     // transposed [N][H][D][L]
    _Float16* Ap = Vt + per;

    dim3 blk(256);
    const int rowTiles = N * L * HEADS / 16;     // 16-row tiles of (n,l,h) rows
    qkv_proj_kernel<false><<<dim3(rowTiles / 8), blk, 0, stream>>>(query, Wq, bq, Qp, N, L);
    qkv_proj_kernel<false><<<dim3(rowTiles / 8), blk, 0, stream>>>(key,   Wk, bk, Kp, N, L);
    qkv_proj_kernel<true ><<<dim3(rowTiles / 8), blk, 0, stream>>>(value, Wv, bv, Vt, N, L);

    const int attnWaves = N * HEADS * (L / 16);
    attn_kernel<<<dim3(attnWaves / 8), blk, 0, stream>>>(Qp, Kp, Vt, mask, Ap, N, L);

    out_proj_kernel<<<dim3(EMBED / 64, (N * L) / 128), blk, 0, stream>>>(Ap, Wo, bo, out);
}
